// GraphAttentionEmbedding_41171556500202
// MI455X (gfx1250) — matrix operand order repaired and checked by reading, not compiled, not run on hardware
//
#include <hip/hip_runtime.h>
#include <hip/hip_bf16.h>
#include <math.h>

typedef __attribute__((ext_vector_type(16))) _Float16 v16h;
typedef __attribute__((ext_vector_type(8)))  _Float16 v8h;
typedef __attribute__((ext_vector_type(8)))  float    v8f;

#define IN_CH    100
#define OUT_CH   100
#define HEADS    2
#define HC       200   // HEADS*OUT_CH
#define EDGE_DIM 200
#define NTILE_HC 13    // ceil(200/16)
#define KC_NODE  4     // ceil(100/32) -> K padded to 128
#define KC_EDGE  7     // ceil(200/32) -> K padded to 224
#define MR       32    // M rows per block (2 x 16-row WMMA subtiles)

// ---- monotonic float <-> uint mapping for order-independent atomic max ----
static __device__ __forceinline__ unsigned floatFlip(float f) {
    unsigned u = __float_as_uint(f);
    return (u & 0x80000000u) ? ~u : (u | 0x80000000u);
}
static __device__ __forceinline__ float floatUnflip(unsigned u) {
    unsigned v = (u & 0x80000000u) ? (u & 0x7FFFFFFFu) : ~u;
    return __uint_as_float(v);
}

// Load a 16x32 f16 A fragment for this lane from an LDS row-major tile.
// Per-lane A data is one contiguous 16-half run (matches packed B order):
//   k in [kc*32 + grp*16, +16)   -> 2x ds_load_b128
static __device__ __forceinline__ v16h load_a_frag(const _Float16* row, int kc, int grp) {
    const _Float16* p = row + kc * 32 + grp * 16;
    v8h lo = *(const v8h*)(p);
    v8h hi = *(const v8h*)(p + 8);
    v16h a;
    #pragma unroll
    for (int i = 0; i < 8; ++i) { a[i] = lo[i]; a[8 + i] = hi[i]; }
    return a;
}

// ============ Prep: pack weights into per-lane WMMA B-fragment order =======
// Fragment (tile, kc): lane -> grp=lane>>4, n=lane&15 ; element e in 0..15:
//   value = W[kc*32 + grp*16 + e][tile*16 + n]   (zero-padded), as f16.
__global__ void pack_node_weights(const float* __restrict__ Wq,
                                  const float* __restrict__ Wk,
                                  const float* __restrict__ Wv,
                                  const float* __restrict__ Ws,
                                  _Float16* __restrict__ Bpk)
{
    int idx = blockIdx.x * blockDim.x + threadIdx.x;
    const int total = 4 * NTILE_HC * KC_NODE * 32 * 16;
    if (idx >= total) return;
    int e    = idx & 15;
    int lane = (idx >> 4) & 31;
    int frag = idx >> 9;               // ((mat*13 + tile)*KC_NODE + kc)
    int kc   = frag % KC_NODE;
    int mt   = frag / KC_NODE;
    int tile = mt % NTILE_HC;
    int mat  = mt / NTILE_HC;
    const float* W = (mat == 0) ? Wq : (mat == 1) ? Wk : (mat == 2) ? Wv : Ws;
    int k   = kc * 32 + (lane >> 4) * 16 + e;
    int col = tile * 16 + (lane & 15);
    float v = (k < IN_CH && col < HC) ? W[(size_t)k * HC + col] : 0.0f;
    Bpk[idx] = (_Float16)v;
}

__global__ void pack_edge_weights(const float* __restrict__ We,
                                  _Float16* __restrict__ Bpk)
{
    int idx = blockIdx.x * blockDim.x + threadIdx.x;
    const int total = NTILE_HC * KC_EDGE * 32 * 16;
    if (idx >= total) return;
    int e    = idx & 15;
    int lane = (idx >> 4) & 31;
    int frag = idx >> 9;               // tile*KC_EDGE + kc
    int kc   = frag % KC_EDGE;
    int tile = frag / KC_EDGE;
    int k   = kc * 32 + (lane >> 4) * 16 + e;
    int col = tile * 16 + (lane & 15);
    float v = (k < EDGE_DIM && col < HC) ? We[(size_t)k * HC + col] : 0.0f;
    Bpk[idx] = (_Float16)v;
}

// ================= Kernel 1: node linears (Q,K,V,skip) via WMMA ============
template<bool GUARD>
static __device__ __forceinline__ void store_rows(float* __restrict__ O, int col,
                                                  int row0, int grp, float bias_v,
                                                  const v8f& acc0, const v8f& acc1, int N)
{
    float* base = O + col;
    #pragma unroll
    for (int r = 0; r < 8; ++r) {
        int m = r + grp * 8;           // D: lanes16-31 hold M=r+8
        int ra = row0 + m, rb = row0 + 16 + m;
        if (!GUARD || ra < N) base[(size_t)ra * HC] = acc0[r] + bias_v;
        if (!GUARD || rb < N) base[(size_t)rb * HC] = acc1[r] + bias_v;
    }
}

__global__ void __launch_bounds__(256)
node_linear_wmma(const float* __restrict__ x,
                 const _Float16* __restrict__ Bpk,   // packed Wq|Wk|Wv|Ws
                 const float* __restrict__ bq, const float* __restrict__ bk,
                 const float* __restrict__ bv, const float* __restrict__ bs,
                 float* __restrict__ Qb, float* __restrict__ Kb,
                 float* __restrict__ Vb, float* __restrict__ outSkip, int N)
{
    __shared__ __align__(16) _Float16 As[MR][128];
    const int tid  = threadIdx.x;
    const int row0 = blockIdx.x * MR;
    const bool full = (row0 + MR <= N);

    for (int idx = tid; idx < MR * 128; idx += 256) {
        int r = idx >> 7, k = idx & 127;
        int row = row0 + r;
        float v = (k < IN_CH && row < N) ? x[(size_t)row * IN_CH + k] : 0.0f;
        As[r][k] = (_Float16)v;
    }
    __syncthreads();

    const int wave = tid >> 5;
    const int lane = tid & 31;
    const int grp  = lane >> 4;
    const int nloc = lane & 15;

    for (int task = wave; task < 4 * NTILE_HC; task += 8) {
        const int mat  = task / NTILE_HC;
        const int tile = task % NTILE_HC;
        const float* bias; float* O;
        switch (mat) {
            case 0:  bias = bq; O = Qb;      break;
            case 1:  bias = bk; O = Kb;      break;
            case 2:  bias = bv; O = Vb;      break;
            default: bias = bs; O = outSkip; break;
        }
        const int col = tile * 16 + nloc;
        const v16h* bfrag = (const v16h*)(Bpk + ((size_t)task * KC_NODE * 32 + lane) * 16);

        v8f acc0 = {}, acc1 = {};
        #pragma unroll
        for (int kc = 0; kc < KC_NODE; ++kc) {
            v16h b  = bfrag[kc * 32];          // 1 coalesced 32B load per lane
            v16h a0 = load_a_frag(&As[nloc][0],      kc, grp);
            v16h a1 = load_a_frag(&As[16 + nloc][0], kc, grp);
            acc0 = __builtin_amdgcn_wmma_f32_16x16x32_f16(
                       false, a0, false, b, (short)0, acc0, false, false);
            acc1 = __builtin_amdgcn_wmma_f32_16x16x32_f16(
                       false, a1, false, b, (short)0, acc1, false, false);
        }
        if (col < HC) {
            float bias_v = bias[col];
            if (full) store_rows<false>(O, col, row0, grp, bias_v, acc0, acc1, N);
            else      store_rows<true >(O, col, row0, grp, bias_v, acc0, acc1, N);
        }
    }
}

// ====== Kernel 2: edge_attr @ We via WMMA, fused k/v add + q.k reduce ======
template<bool GUARD>
static __device__ __forceinline__ void edge_epilogue(
        const v8f& acc, int e0, int grp, int nloc, int col, bool colValid,
        int head, int hlo, int hhi,
        const int* sSrc, const int* sDst,
        const float* __restrict__ Qb, const float* __restrict__ Kb,
        const float* __restrict__ Vb,
        float* __restrict__ Vfull, float* __restrict__ alpha, int E)
{
    #pragma unroll
    for (int r = 0; r < 8; ++r) {
        const int m = r + grp * 8;           // edge row within 16-row subtile
        const int e = e0 + m;
        const bool ev = !GUARD || (e < E);
        float prod = 0.0f;
        if (colValid && ev) {
            const int sv = sSrc[m], dv = sDst[m];
            float em = acc[r];
            float vf = em + Vb[(size_t)sv * HC + col];
            Vfull[(size_t)e * HC + col] = vf;
            float kf = em + Kb[(size_t)sv * HC + col];
            float qf = Qb[(size_t)dv * HC + col];
            prod = qf * kf * 0.1f;           // 1/sqrt(100)
        }
        for (int h = hlo; h <= hhi; ++h) {
            float p = (head == h) ? prod : 0.0f;
            p += __shfl_xor(p, 1, 16);
            p += __shfl_xor(p, 2, 16);
            p += __shfl_xor(p, 4, 16);
            p += __shfl_xor(p, 8, 16);
            if (nloc == 0 && ev) atomicAdd(&alpha[(size_t)e * 2 + h], p);
        }
    }
}

__global__ void __launch_bounds__(256)
edge_attn_wmma(const float* __restrict__ last_update,
               const int*   __restrict__ edge_index,
               const float* __restrict__ t,
               const float* __restrict__ msg,
               const float* __restrict__ w_time,
               const float* __restrict__ b_time,
               const _Float16* __restrict__ Bpk,    // packed We
               const float* __restrict__ Qb,
               const float* __restrict__ Kb,
               const float* __restrict__ Vb,
               float* __restrict__ Vfull,
               float* __restrict__ alpha,
               int E)
{
    __shared__ __align__(16) _Float16 As[MR][224];  // K padded 200 -> 224
    __shared__ int   sSrc[MR], sDst[MR];
    __shared__ float sRel[MR];
    const int tid = threadIdx.x;
    const int e0  = blockIdx.x * MR;
    const bool full = (e0 + MR <= E);

    if (tid < MR) {
        int e = e0 + tid;
        int s = 0, d = 0; float rel = 0.0f;
        if (e < E) {
            s = edge_index[e];
            d = edge_index[E + e];
            rel = fabsf(last_update[s] - t[e]);
        }
        sSrc[tid] = s; sDst[tid] = d; sRel[tid] = rel;
    }
    __syncthreads();

    for (int idx = tid; idx < MR * 224; idx += 256) {
        int r = idx / 224, k = idx % 224;
        int e = e0 + r;
        float v = 0.0f;
        if (e < E) {
            if (k < 100)      v = __cosf(sRel[r] * w_time[k] + b_time[k]);
            else if (k < 200) v = msg[(size_t)e * 100 + (k - 100)];
        }
        As[r][k] = (_Float16)v;
    }
    __syncthreads();

    const int wave = tid >> 5;
    const int lane = tid & 31;
    const int grp  = lane >> 4;
    const int nloc = lane & 15;

    for (int tile = wave; tile < NTILE_HC; tile += 8) {
        const int col = tile * 16 + nloc;
        const bool colValid = (col < HC);
        const int head = (col >= OUT_CH) ? 1 : 0;
        const int hlo = (tile * 16 < OUT_CH) ? 0 : 1;
        const int hhi = (tile * 16 + 15 >= OUT_CH) ? 1 : 0;
        const v16h* bfrag = (const v16h*)(Bpk + ((size_t)tile * KC_EDGE * 32 + lane) * 16);

        v8f acc0 = {}, acc1 = {};
        #pragma unroll
        for (int kc = 0; kc < KC_EDGE; ++kc) {
            v16h b  = bfrag[kc * 32];
            v16h a0 = load_a_frag(&As[nloc][0],      kc, grp);
            v16h a1 = load_a_frag(&As[16 + nloc][0], kc, grp);
            acc0 = __builtin_amdgcn_wmma_f32_16x16x32_f16(
                       false, a0, false, b, (short)0, acc0, false, false);
            acc1 = __builtin_amdgcn_wmma_f32_16x16x32_f16(
                       false, a1, false, b, (short)0, acc1, false, false);
        }

        if (full) {
            edge_epilogue<false>(acc0, e0,      grp, nloc, col, colValid, head, hlo, hhi,
                                 sSrc,      sDst,      Qb, Kb, Vb, Vfull, alpha, E);
            edge_epilogue<false>(acc1, e0 + 16, grp, nloc, col, colValid, head, hlo, hhi,
                                 sSrc + 16, sDst + 16, Qb, Kb, Vb, Vfull, alpha, E);
        } else {
            edge_epilogue<true>(acc0, e0,      grp, nloc, col, colValid, head, hlo, hhi,
                                sSrc,      sDst,      Qb, Kb, Vb, Vfull, alpha, E);
            edge_epilogue<true>(acc1, e0 + 16, grp, nloc, col, colValid, head, hlo, hhi,
                                sSrc + 16, sDst + 16, Qb, Kb, Vb, Vfull, alpha, E);
        }
    }
}

// ================= Kernel 3: segment max over dst (order-independent) ======
__global__ void seg_max_kernel(const float* __restrict__ alpha,
                               const int* __restrict__ edge_index,
                               unsigned* __restrict__ amaxkey, int E)
{
    int i = blockIdx.x * blockDim.x + threadIdx.x;
    if (i >= 2 * E) return;
    int e = i >> 1, h = i & 1;
    int d = edge_index[E + e];
    atomicMax(&amaxkey[(size_t)d * 2 + h], floatFlip(alpha[i]));
}

// ================= Kernel 4: exp + segment sum =============================
__global__ void seg_expsum_kernel(float* __restrict__ alpha,
                                  const int* __restrict__ edge_index,
                                  const unsigned* __restrict__ amaxkey,
                                  float* __restrict__ denom, int E)
{
    int i = blockIdx.x * blockDim.x + threadIdx.x;
    if (i >= 2 * E) return;
    int e = i >> 1, h = i & 1;
    int d = edge_index[E + e];
    float ex = __expf(alpha[i] - floatUnflip(amaxkey[(size_t)d * 2 + h]));
    alpha[i] = ex;                              // reuse buffer for exp values
    atomicAdd(&denom[(size_t)d * 2 + h], ex);
}

// ================= Kernel 5: weighted scatter of v =========================
__global__ void aggregate_kernel(const float* __restrict__ exv,
                                 const int* __restrict__ edge_index,
                                 const float* __restrict__ denom,
                                 const float* __restrict__ Vfull,
                                 float* __restrict__ out, int E)
{
    size_t i = (size_t)blockIdx.x * blockDim.x + threadIdx.x;
    if (i >= (size_t)E * HC) return;
    int e = (int)(i / HC);
    int c = (int)(i % HC);
    int h = (c >= OUT_CH) ? 1 : 0;
    int d = edge_index[E + e];
    float attn = exv[(size_t)e * 2 + h] / (denom[(size_t)d * 2 + h] + 1e-16f);
    atomicAdd(&out[(size_t)d * HC + c], Vfull[i] * attn);
}

// ===========================================================================
extern "C" void kernel_launch(void* const* d_in, const int* in_sizes, int n_in,
                              void* d_out, int out_size, void* d_ws, size_t ws_size,
                              hipStream_t stream)
{
    const float* x           = (const float*)d_in[0];
    const float* last_update = (const float*)d_in[1];
    const int*   edge_index  = (const int*)  d_in[2];
    const float* t           = (const float*)d_in[3];
    const float* msg         = (const float*)d_in[4];
    const float* w_time      = (const float*)d_in[5];
    const float* b_time      = (const float*)d_in[6];
    const float* Wq = (const float*)d_in[7];
    const float* bq = (const float*)d_in[8];
    const float* Wk = (const float*)d_in[9];
    const float* bk = (const float*)d_in[10];
    const float* Wv = (const float*)d_in[11];
    const float* bv = (const float*)d_in[12];
    const float* We = (const float*)d_in[13];
    const float* Ws = (const float*)d_in[14];
    const float* bs = (const float*)d_in[15];

    const int N = in_sizes[1];   // last_update: [N]
    const int E = in_sizes[3];   // t: [E]
    float* out = (float*)d_out;

    // Workspace layout (all offsets 32B-aligned)
    char* p = (char*)d_ws;
    float* Qb    = (float*)p; p += (size_t)N * HC * sizeof(float);
    float* Kb    = (float*)p; p += (size_t)N * HC * sizeof(float);
    float* Vb    = (float*)p; p += (size_t)N * HC * sizeof(float);
    float* Vfull = (float*)p; p += (size_t)E * HC * sizeof(float);
    float* alpha = (float*)p; p += (size_t)E * 2  * sizeof(float);
    float* denom = (float*)p; p += (size_t)N * 2  * sizeof(float);
    unsigned* amaxkey = (unsigned*)p; p += (size_t)N * 2 * sizeof(unsigned);
    _Float16* BpkNode = (_Float16*)p; p += (size_t)4 * NTILE_HC * KC_NODE * 512 * sizeof(_Float16);
    _Float16* BpkEdge = (_Float16*)p;

    hipMemsetAsync(alpha,   0, (size_t)E * 2 * sizeof(float),    stream);
    hipMemsetAsync(denom,   0, (size_t)N * 2 * sizeof(float),    stream);
    hipMemsetAsync(amaxkey, 0, (size_t)N * 2 * sizeof(unsigned), stream);

    {   // pack weights into WMMA B-fragment order (f16, zero-padded)
        int tot1 = 4 * NTILE_HC * KC_NODE * 512;
        int tot2 = NTILE_HC * KC_EDGE * 512;
        pack_node_weights<<<(tot1 + 255) / 256, 256, 0, stream>>>(Wq, Wk, Wv, Ws, BpkNode);
        pack_edge_weights<<<(tot2 + 255) / 256, 256, 0, stream>>>(We, BpkEdge);
    }

    node_linear_wmma<<<(N + MR - 1) / MR, 256, 0, stream>>>(
        x, BpkNode, bq, bk, bv, bs, Qb, Kb, Vb, out, N);

    edge_attn_wmma<<<(E + MR - 1) / MR, 256, 0, stream>>>(
        last_update, edge_index, t, msg, w_time, b_time, BpkEdge,
        Qb, Kb, Vb, Vfull, alpha, E);

    seg_max_kernel<<<(2 * E + 255) / 256, 256, 0, stream>>>(
        alpha, edge_index, amaxkey, E);

    seg_expsum_kernel<<<(2 * E + 255) / 256, 256, 0, stream>>>(
        alpha, edge_index, amaxkey, denom, E);

    size_t total = (size_t)E * HC;
    aggregate_kernel<<<(unsigned)((total + 255) / 256), 256, 0, stream>>>(
        alpha, edge_index, denom, Vfull, out, E);
}
